// FFOsc_59347858096498
// MI455X (gfx1250) — matrix operand order repaired and checked by reading, not compiled, not run on hardware
//
#include <hip/hip_runtime.h>
#include <hip/hip_bf16.h>

typedef __bf16 bf16;
typedef __attribute__((ext_vector_type(16))) __bf16 v16bf;
typedef __attribute__((ext_vector_type(8)))  float  v8f;

#define C_CELLS 4
#define Bsz 64
#define Tsz 512
#define Dsz 256
#define Usz 512
#define OSC 5
#define KTOT 768            // D + U (fused [x_t | h] GEMM)
#define NKT  (KTOT / 32)    // 24 K-steps of 32
#define NKX  (Dsz / 32)     // 8 K-steps from x
#define NKH  (Usz / 32)     // 16 K-steps from h
#define BT   16             // batch rows per workgroup

// ---- workspace layout (bytes) ----
#define OFF_X    0
#define SZ_X     (Tsz * Bsz * Dsz * 2)          // x as bf16, [T][B][D]  (16 MB)
#define OFF_WT   (OFF_X + SZ_X)
#define SZ_WT    (C_CELLS * 2 * KTOT * Usz * 2) // tiled bf16 weights     (6 MB)
#define OFF_BIAS (OFF_WT + SZ_WT)
#define SZ_BIAS  (C_CELLS * 2 * Usz * 4)        // combined bias f32
#define OFF_SEQ  (OFF_BIAS + SZ_BIAS)           // seq [T][C][B][U] f32  (256 MB)

// ---------------- prep: x (f32 [B][T][D]) -> bf16 [T][B][D] ----------------
__global__ void FFOsc_prep_x(const float* __restrict__ x, bf16* __restrict__ xw) {
    int tid = blockIdx.x * blockDim.x + threadIdx.x;
    int d = tid % Dsz;
    int b = (tid / Dsz) % Bsz;
    int t = tid / (Dsz * Bsz);
    if (t >= Tsz) return;
    xw[(t * Bsz + b) * Dsz + d] = (bf16)x[(b * Tsz + t) * Dsz + d];
}

// ---- prep: weights -> WMMA-B tiled bf16 layout -------------------------------
// WT index = ((((cell*2+gate)*32 + ntile)*NKT + kt)*32 + lane)*16 + i
// element maps to K = kt*32 + (lane/16)*8 + i (i<8), kt*32 + 16 + (lane/16)*8 + (i-8) (i>=8) ; u = ntile*16 + (lane%16)
// K < 256 -> Wx row K ; K >= 256 -> Wh row K-256  (fused [x|h] K dimension)
__global__ void FFOsc_prep_w(const float* __restrict__ Wjx, const float* __restrict__ Wjh,
                             const float* __restrict__ Wkx, const float* __restrict__ Wkh,
                             bf16* __restrict__ wt) {
    int tid = blockIdx.x * blockDim.x + threadIdx.x;
    int i    = tid & 15;
    int lane = (tid >> 4) & 31;
    int kt   = (tid >> 9) % NKT;
    int ntl  = (tid / (512 * NKT)) & 31;
    int gate = (tid / (512 * NKT * 32)) & 1;
    int cell = tid / (512 * NKT * 32 * 2);
    if (cell >= C_CELLS) return;
    int K = kt * 32 + (lane >> 4) * 8 + (i & 7) + ((i >> 3) << 4);
    int u = ntl * 16 + (lane & 15);
    float v;
    if (gate == 0)
        v = (K < Dsz) ? Wjx[(cell * Dsz + K) * Usz + u] : Wjh[(cell * Usz + (K - Dsz)) * Usz + u];
    else
        v = (K < Dsz) ? Wkx[(cell * Dsz + K) * Usz + u] : Wkh[(cell * Usz + (K - Dsz)) * Usz + u];
    wt[tid] = (bf16)v;
}

// ---- prep: bias[cell][gate][u] = b{j,k}x + b{j,k}h ---------------------------
__global__ void FFOsc_prep_bias(const float* __restrict__ bjx, const float* __restrict__ bjh,
                                const float* __restrict__ bkx, const float* __restrict__ bkh,
                                float* __restrict__ bias) {
    int tid = blockIdx.x * blockDim.x + threadIdx.x;
    if (tid >= C_CELLS * 2 * Usz) return;
    int u = tid & (Usz - 1);
    int gate = (tid >> 9) & 1;
    int cell = tid >> 10;
    bias[tid] = (gate == 0) ? (bjx[cell * Usz + u] + bjh[cell * Usz + u])
                            : (bkx[cell * Usz + u] + bkh[cell * Usz + u]);
}

// ---- fused flip-flop scan: 1 WG = (cell, 16 batch rows); WMMA recurrence -----
__global__ __launch_bounds__(512, 1)
void FFOsc_scan(const float* __restrict__ h0,
                const bf16*  __restrict__ xw,
                const bf16*  __restrict__ wt,
                const float* __restrict__ bias,
                float*       __restrict__ seq) {
    __shared__ __align__(16) bf16  s_x[2][BT * Dsz];    // 16 KB  x_t double buffer (bf16)
    __shared__ __align__(16) bf16  s_h16[BT * Usz];     // 16 KB  h (bf16 copy for WMMA A)
    __shared__ __align__(16) float s_h[BT * Usz];       // 32 KB  h master (f32)
    __shared__ __align__(16) float s_jk[2][BT * Usz];   // 64 KB  sigmoid(j), sigmoid(k)
    __shared__               float s_bias[2 * Usz];     //  4 KB

    const int cell = blockIdx.x & 3;
    const int b0   = (blockIdx.x >> 2) * BT;
    const int tid  = threadIdx.x;
    const int lane = tid & 31;
    const int w    = tid >> 5;     // wave 0..15
    const int q16  = lane >> 4;    // lane half
    const int m    = lane & 15;

    // init h state + bias cache
    for (int idx = tid; idx < BT * Usz; idx += 512) {
        int mm = idx >> 9, u = idx & (Usz - 1);
        float hv = h0[(cell * Bsz + b0 + mm) * Usz + u];
        s_h[idx]   = hv;
        s_h16[idx] = (bf16)hv;
    }
    for (int idx = tid; idx < 2 * Usz; idx += 512)
        s_bias[idx] = bias[cell * (2 * Usz) + idx];

    const bf16* wt_cell = wt + cell * (2 * KTOT * Usz);   // this cell's tiled weights

    // async Global->LDS stage of x_t tile: 512 threads x 16B = 8 KB, ASYNCcnt-tracked
    const unsigned lds_x0   = (unsigned)(uintptr_t)(&s_x[0][0]);
    const unsigned lds_x1   = (unsigned)(uintptr_t)(&s_x[1][0]);
    const unsigned voff     = (unsigned)tid * 16u;
    #define STAGE_ASYNC(T, BUF)                                                     \
        {                                                                           \
            const bf16* gsrc = xw + ((T) * Bsz + b0) * Dsz;                         \
            unsigned lds = ((BUF) ? lds_x1 : lds_x0) + voff;                        \
            asm volatile("global_load_async_to_lds_b128 %0, %1, %2"                 \
                         :: "v"(lds), "v"(voff), "s"(gsrc) : "memory");             \
        }

    STAGE_ASYNC(0, 0)
    __syncthreads();

    for (int t = 0; t < Tsz; ++t) {
        const int buf = t & 1;
        asm volatile("s_wait_asynccnt 0x0" ::: "memory");
        __syncthreads();                       // x_t staged; h from step t-1 visible
        if (t + 1 < Tsz) STAGE_ASYNC(t + 1, buf ^ 1)

        v8f acc[4];
        #pragma unroll
        for (int q = 0; q < 4; ++q) acc[q] = (v8f){0.f,0.f,0.f,0.f,0.f,0.f,0.f,0.f};

        // ---- K-part 1: x_t (K = 0..255), branchless A loads from s_x[buf] ----
        {
            const uint4* ra = (const uint4*)(s_x[buf] + m * Dsz);
            for (int kt = 0; kt < NKX; ++kt) {
                union { uint4 u4[2]; v16bf v; } a;
                a.u4[0] = ra[kt * 4 + q16];        // elems 0..7  : K = kt*32 + q16*8 + j
                a.u4[1] = ra[kt * 4 + 2 + q16];    // elems 8..15 : K = kt*32+16 + q16*8 + j
                #pragma unroll
                for (int q = 0; q < 4; ++q) {
                    const int nt = w * 4 + q;      // 0..63 : gate = nt>>5, col tile = nt&31
                    union { uint4 u4[2]; v16bf v; } b;
                    const uint4* bp = (const uint4*)(wt_cell + (((nt * NKT + kt) * 32 + lane) << 4));
                    b.u4[0] = bp[0];
                    b.u4[1] = bp[1];
                    acc[q] = __builtin_amdgcn_wmma_f32_16x16x32_bf16(
                        false, a.v, false, b.v, (short)0, acc[q], false, false);
                }
            }
        }
        // ---- K-part 2: h (K = 256..767), branchless A loads from s_h16 ----
        {
            const uint4* ra = (const uint4*)(s_h16 + m * Usz);
            for (int kh = 0; kh < NKH; ++kh) {
                union { uint4 u4[2]; v16bf v; } a;
                a.u4[0] = ra[kh * 4 + q16];
                a.u4[1] = ra[kh * 4 + 2 + q16];
                const int kt = NKX + kh;
                #pragma unroll
                for (int q = 0; q < 4; ++q) {
                    const int nt = w * 4 + q;
                    union { uint4 u4[2]; v16bf v; } b;
                    const uint4* bp = (const uint4*)(wt_cell + (((nt * NKT + kt) * 32 + lane) << 4));
                    b.u4[0] = bp[0];
                    b.u4[1] = bp[1];
                    acc[q] = __builtin_amdgcn_wmma_f32_16x16x32_bf16(
                        false, a.v, false, b.v, (short)0, acc[q], false, false);
                }
            }
        }

        // bias + sigmoid, scatter to LDS per C/D layout (M = v + 8*half, N = lane%16)
        #pragma unroll
        for (int q = 0; q < 4; ++q) {
            const int nt   = w * 4 + q;
            const int gate = nt >> 5;
            const int col  = (nt & 31) * 16 + m;
            const float bb = s_bias[gate * Usz + col];
            #pragma unroll
            for (int v = 0; v < 8; ++v) {
                const int M = v + 8 * q16;
                const float pre = acc[q][v] + bb;
                s_jk[gate][M * Usz + col] = 1.0f / (1.0f + __expf(-pre));
            }
        }
        __syncthreads();

        // elementwise flip-flop update + write seq[t][cell][b][u]
        // (update -> next-step GEMM hazard is covered by next iteration's top barrier)
        float* seq_t = seq + ((t * C_CELLS + cell) * Bsz + b0) * Usz;
        #pragma unroll
        for (int mm = 0; mm < BT; ++mm) {
            const int idx = mm * Usz + tid;
            const float jv = s_jk[0][idx];
            const float kv = s_jk[1][idx];
            const float hv = s_h[idx];
            const float hn = jv * (1.0f - hv) + (1.0f - kv) * hv;
            s_h[idx]   = hn;
            s_h16[idx] = (bf16)hn;
            seq_t[idx] = hn;
        }
        __syncthreads();
    }
    #undef STAGE_ASYNC
}

// ---- oscillator unroll: seq[T][C][B][U] -> out[B][T*5][U] --------------------
__global__ void FFOsc_osc(const float* __restrict__ seq, float* __restrict__ out) {
    int tid = blockIdx.x * blockDim.x + threadIdx.x;
    if (tid >= Bsz * Tsz * Usz) return;
    const int u = tid % Usz;
    const int t = (tid / Usz) % Tsz;
    const int b = tid / (Usz * Tsz);
    const int base = (t * C_CELLS) * (Bsz * Usz) + b * Usz + u;
    const float phi   = seq[base + 0 * Bsz * Usz];
    const float omega = seq[base + 1 * Bsz * Usz];
    const float r     = seq[base + 2 * Bsz * Usz];
    const float mu    = seq[base + 3 * Bsz * Usz];
    const long ob = ((long)b * (Tsz * OSC) + t * OSC) * Usz + u;
    out[ob] = r * __cosf(phi);
    float rt = r, ph = phi;
    #pragma unroll
    for (int s = 1; s < OSC; ++s) {
        rt = rt + (mu - rt * rt) * rt;
        ph = ph + omega;
        out[ob + (long)s * Usz] = rt * __cosf(ph);
    }
}

extern "C" void kernel_launch(void* const* d_in, const int* in_sizes, int n_in,
                              void* d_out, int out_size, void* d_ws, size_t ws_size,
                              hipStream_t stream) {
    const float* x   = (const float*)d_in[0];
    const float* h0  = (const float*)d_in[1];
    const float* Wjx = (const float*)d_in[2];
    const float* bjx = (const float*)d_in[3];
    const float* Wjh = (const float*)d_in[4];
    const float* bjh = (const float*)d_in[5];
    const float* Wkx = (const float*)d_in[6];
    const float* bkx = (const float*)d_in[7];
    const float* Wkh = (const float*)d_in[8];
    const float* bkh = (const float*)d_in[9];
    float* out = (float*)d_out;

    char* ws = (char*)d_ws;
    bf16*  xw   = (bf16*)(ws + OFF_X);
    bf16*  wt   = (bf16*)(ws + OFF_WT);
    float* bias = (float*)(ws + OFF_BIAS);
    float* seq  = (float*)(ws + OFF_SEQ);

    FFOsc_prep_x   <<<(Tsz * Bsz * Dsz) / 256, 256, 0, stream>>>(x, xw);
    FFOsc_prep_w   <<<(C_CELLS * 2 * KTOT * Usz) / 256, 256, 0, stream>>>(Wjx, Wjh, Wkx, Wkh, wt);
    FFOsc_prep_bias<<<(C_CELLS * 2 * Usz + 255) / 256, 256, 0, stream>>>(bjx, bjh, bkx, bkh, bias);
    FFOsc_scan     <<<C_CELLS * (Bsz / BT), 512, 0, stream>>>(h0, xw, wt, bias, seq);
    FFOsc_osc      <<<(Bsz * Tsz * Usz) / 256, 256, 0, stream>>>(seq, out);
}
